// GATv2Conv_module_44719199486484
// MI455X (gfx1250) — compile-verified, matrix operand order checked by reading, not run on hardware
//
#include <hip/hip_runtime.h>
#include <cmath>

#define N_NODES 50000
#define N_EDGES 640000
#define DIM     128
#define EDIM    64
#define HEADS   4
#define CH      32
#define EN      (N_EDGES + N_NODES)   // 690000
#define MID_DIM 256

typedef __attribute__((ext_vector_type(16))) __bf16 v16bf;
typedef __attribute__((ext_vector_type(8)))  __bf16 v8bf;
typedef __attribute__((ext_vector_type(8)))  float  v8f;
typedef __attribute__((ext_vector_type(4)))  float  v4f;

// ---------------- workspace layout (bytes) ----------------
constexpr size_t alup(size_t x) { return (x + 255) & ~(size_t)255; }
constexpr size_t OFF_DEG  = 0;
constexpr size_t OFF_LOOP = OFF_DEG  + alup((size_t)N_NODES * 4);
constexpr size_t OFF_XL   = OFF_LOOP + alup((size_t)N_NODES * EDIM * 4);
constexpr size_t OFF_XR   = OFF_XL   + alup((size_t)N_NODES * DIM * 4);
constexpr size_t OFF_GAT  = OFF_XR   + alup((size_t)N_NODES * DIM * 4);
constexpr size_t OFF_H    = OFF_GAT  + alup((size_t)N_NODES * DIM * 4);
constexpr size_t OFF_MID  = OFF_H    + alup((size_t)N_NODES * DIM * 4);
constexpr size_t OFF_MLP  = OFF_MID  + alup((size_t)N_NODES * MID_DIM * 4);
constexpr size_t OFF_LOG  = OFF_MLP  + alup((size_t)N_NODES * DIM * 4);
constexpr size_t OFF_AMAX = OFF_LOG  + alup((size_t)EN * HEADS * 4);
constexpr size_t OFF_DEN  = OFF_AMAX + alup((size_t)N_NODES * HEADS * 4);
constexpr size_t OFF_WL16 = OFF_DEN  + alup((size_t)N_NODES * HEADS * 4);
constexpr size_t OFF_WR16 = OFF_WL16 + alup((size_t)DIM * DIM * 2);
constexpr size_t OFF_WE16 = OFF_WR16 + alup((size_t)DIM * DIM * 2);
constexpr size_t OFF_W116 = OFF_WE16 + alup((size_t)DIM * EDIM * 2);
constexpr size_t OFF_W216 = OFF_W116 + alup((size_t)MID_DIM * DIM * 2);

// ---------------- fragment helpers ----------------
// 16-bit A/B operand layout: lane (0-15) row/col = lane&15, K chunks at
// kb+0..7 and kb+16..23 where kb = (lane>>4)*8 (relative to the k-tile).
__device__ inline v16bf frag_from_f32(const float* row, int kbase) {
  v4f a0 = *(const v4f*)(row + kbase);
  v4f a1 = *(const v4f*)(row + kbase + 4);
  v4f a2 = *(const v4f*)(row + kbase + 16);
  v4f a3 = *(const v4f*)(row + kbase + 20);
  v16bf f;
#pragma unroll
  for (int i = 0; i < 4; ++i) {
    f[i]      = (__bf16)a0[i];
    f[i + 4]  = (__bf16)a1[i];
    f[i + 8]  = (__bf16)a2[i];
    f[i + 12] = (__bf16)a3[i];
  }
  return f;
}
__device__ inline v16bf frag_from_u16(const unsigned short* row, int kbase) {
  v8bf lo = *(const v8bf*)(row + kbase);        // 16B vector load
  v8bf hi = *(const v8bf*)(row + kbase + 16);   // 16B vector load
  return __builtin_shufflevector(lo, hi, 0, 1, 2, 3, 4, 5, 6, 7,
                                 8, 9, 10, 11, 12, 13, 14, 15);
}
__device__ inline v8f wmma_bf16(v16bf a, v16bf b, v8f c) {
  return __builtin_amdgcn_wmma_f32_16x16x32_bf16(false, a, false, b, (short)0, c,
                                                 false, false);
}
__device__ inline void store8(float* dst, v8f acc) {
  *(v4f*)dst       = (v4f){acc[0], acc[1], acc[2], acc[3]};
  *(v4f*)(dst + 4) = (v4f){acc[4], acc[5], acc[6], acc[7]};
}
__device__ inline void store8_bias(float* dst, const float* bias, v8f acc) {
  v4f b0 = *(const v4f*)bias;
  v4f b1 = *(const v4f*)(bias + 4);
  *(v4f*)dst       = (v4f){acc[0] + b0[0], acc[1] + b0[1], acc[2] + b0[2], acc[3] + b0[3]};
  *(v4f*)(dst + 4) = (v4f){acc[4] + b1[0], acc[5] + b1[1], acc[6] + b1[2], acc[7] + b1[3]};
}
__device__ inline void atomicMaxFloat(float* addr, float v) {
  if (v >= 0.0f) atomicMax((int*)addr, __float_as_int(v));
  else           atomicMin((unsigned int*)addr, __float_as_uint(v));
}

// ---------------- small utility kernels ----------------
__global__ void k_fill(float* p, float v, int n) {
  int i = blockIdx.x * blockDim.x + threadIdx.x;
  if (i < n) p[i] = v;
}
__global__ void k_init_gat(float* gat, const float* bias, int n) {
  int i = blockIdx.x * blockDim.x + threadIdx.x;
  if (i < n) gat[i] = bias[i & (DIM - 1)];
}
__global__ void k_cvt_bf16(const float* s, unsigned short* d, int n) {
  int i = blockIdx.x * blockDim.x + threadIdx.x;
  if (i < n) {
    __bf16 v = (__bf16)s[i];
    d[i] = __builtin_bit_cast(unsigned short, v);
  }
}
__global__ void k_ei(const int* ei, int* ei_out) {
  int r = blockIdx.x * blockDim.x + threadIdx.x;
  if (r >= EN) return;
  int s = (r < N_EDGES) ? ei[r]           : r - N_EDGES;
  int d = (r < N_EDGES) ? ei[N_EDGES + r] : r - N_EDGES;
  ei_out[r]      = s;
  ei_out[EN + r] = d;
}

// ---------------- self-loop attr (mean of incoming edge_attr) ----------------
__global__ void k_loop_accum(const int* ei, const float* ea, float* deg, float* loopb) {
  int t = blockIdx.x * blockDim.x + threadIdx.x;
  if (t >= N_EDGES * 4) return;
  int e = t >> 2, q = t & 3;
  int d = ei[N_EDGES + e];
  if (q == 0) atomicAdd(&deg[d], 1.0f);
  const float* src = ea + (size_t)e * EDIM + q * 16;
  float* dst = loopb + (size_t)d * EDIM + q * 16;
#pragma unroll
  for (int c = 0; c < 16; c += 4) {
    v4f v = *(const v4f*)(src + c);
#pragma unroll
    for (int j = 0; j < 4; ++j) atomicAdd(&dst[c + j], v[j]);
  }
}
__global__ void k_loop_div(const float* deg, float* loopb) {
  int i = blockIdx.x * blockDim.x + threadIdx.x;
  if (i >= N_NODES * EDIM) return;
  loopb[i] = loopb[i] / fmaxf(deg[i >> 6], 1.0f);
}

// ---------------- node projections xl/xr via WMMA ----------------
__global__ __launch_bounds__(128) void k_proj(const float* x,
    const unsigned short* Wl16, const float* bl,
    const unsigned short* Wr16, const float* br, float* xl, float* xr) {
  int wave = threadIdx.x >> 5, lane = threadIdx.x & 31;
  int ntile = blockIdx.x * 4 + wave;
  if (ntile * 16 >= N_NODES) return;
  int node = ntile * 16 + (lane & 15);
  int ksub = (lane >> 4) << 3;
  const float* xrow = x + (size_t)node * DIM;
  v16bf b[4];
#pragma unroll
  for (int kt = 0; kt < 4; ++kt) b[kt] = frag_from_f32(xrow, kt * 32 + ksub);
#pragma unroll
  for (int mt = 0; mt < 8; ++mt) {
    int mrow = mt * 16 + (lane & 15);
    v8f accL = {}, accR = {};
#pragma unroll
    for (int kt = 0; kt < 4; ++kt) {
      accL = wmma_bf16(frag_from_u16(Wl16 + (size_t)mrow * DIM, kt * 32 + ksub), b[kt], accL);
      accR = wmma_bf16(frag_from_u16(Wr16 + (size_t)mrow * DIM, kt * 32 + ksub), b[kt], accR);
    }
    int mbase = mt * 16 + ksub;
    store8_bias(xl + (size_t)node * DIM + mbase, bl + mbase, accL);
    store8_bias(xr + (size_t)node * DIM + mbase, br + mbase, accR);
  }
}

// ---------------- fused edge kernel: e=ea@We^T (WMMA, tile in LDS),
// g=leakyrelu(xl[s]+xr[d]+e), logits=g.att, segment-max ----------------
__global__ __launch_bounds__(128) void k_edge(const int* ei, const float* ea,
    const float* loopb, const unsigned short* We16, const float* xl,
    const float* xr, const float* att, float* logits, float* amax) {
  __shared__ float e_lds[64 * DIM];   // 32 KB: 4 waves x 16 edges x 128 ch
  int wave = threadIdx.x >> 5, lane = threadIdx.x & 31;
  int tile = blockIdx.x * 4 + wave;
  int rowbase = tile * 16;
  if (rowbase < EN) {
    int row = rowbase + (lane & 15);
    const float* rptr = (row < N_EDGES) ? (ea + (size_t)row * EDIM)
                                        : (loopb + (size_t)(row - N_EDGES) * EDIM);
    int ksub = (lane >> 4) << 3;
    v16bf b0 = frag_from_f32(rptr, ksub);
    v16bf b1 = frag_from_f32(rptr, 32 + ksub);
    int el = wave * 16 + (lane & 15);
#pragma unroll
    for (int mt = 0; mt < 8; ++mt) {
      int mrow = mt * 16 + (lane & 15);
      v8f acc = {};
      acc = wmma_bf16(frag_from_u16(We16 + (size_t)mrow * EDIM, ksub),      b0, acc);
      acc = wmma_bf16(frag_from_u16(We16 + (size_t)mrow * EDIM, 32 + ksub), b1, acc);
      store8(&e_lds[el * DIM + mt * 16 + ksub], acc);
    }
  }
  __syncthreads();
  int blockrow = blockIdx.x * 64;
  for (int p = threadIdx.x; p < 256; p += 128) {   // 64 edges x 4 heads
    int el = p >> 2, hh = p & 3;
    int row = blockrow + el;
    if (row >= EN) continue;
    int s, d;
    if (row < N_EDGES) { s = ei[row]; d = ei[N_EDGES + row]; }
    else               { s = d = row - N_EDGES; }
    const float* xls = xl + (size_t)s * DIM + hh * CH;
    const float* xrd = xr + (size_t)d * DIM + hh * CH;
    const float* ee  = &e_lds[el * DIM + hh * CH];
    const float* av  = att + hh * CH;
    float lg = 0.f;
#pragma unroll
    for (int c = 0; c < CH; c += 4) {
      v4f xa = *(const v4f*)(xls + c);
      v4f xb = *(const v4f*)(xrd + c);
      v4f ec = *(const v4f*)(ee + c);
      v4f aw = *(const v4f*)(av + c);
#pragma unroll
      for (int j = 0; j < 4; ++j) {
        float g = xa[j] + xb[j] + ec[j];
        g = (g > 0.f) ? g : 0.2f * g;
        lg += aw[j] * g;
      }
    }
    logits[(size_t)row * HEADS + hh] = lg;
    atomicMaxFloat(&amax[(size_t)d * HEADS + hh], lg);
  }
}

// ---------------- segment softmax ----------------
__global__ void k_softmax(const int* ei, const float* logits, const float* amax,
                          float* denom, float* alpha_out) {
  int idx = blockIdx.x * blockDim.x + threadIdx.x;
  if (idx >= EN * HEADS) return;
  int row = idx >> 2, hh = idx & 3;
  int d = (row < N_EDGES) ? ei[N_EDGES + row] : row - N_EDGES;
  float ex = expf(logits[idx] - amax[(size_t)d * HEADS + hh]);
  alpha_out[idx] = ex;
  atomicAdd(&denom[(size_t)d * HEADS + hh], ex);
}
__global__ void k_agg(const int* ei, const float* denom, const float* xl,
                      float* alpha_out, float* gat) {
  int idx = blockIdx.x * blockDim.x + threadIdx.x;
  if (idx >= EN * HEADS) return;
  int row = idx >> 2, hh = idx & 3;
  int s, d;
  if (row < N_EDGES) { s = ei[row]; d = ei[N_EDGES + row]; }
  else               { s = d = row - N_EDGES; }
  float alpha = alpha_out[idx] / denom[(size_t)d * HEADS + hh];
  alpha_out[idx] = alpha;
  const float* xs = xl + (size_t)s * DIM + hh * CH;
  float* gd = gat + (size_t)d * DIM + hh * CH;
#pragma unroll
  for (int c = 0; c < CH; c += 4) {
    v4f xv = *(const v4f*)(xs + c);
#pragma unroll
    for (int j = 0; j < 4; ++j) atomicAdd(&gd[c + j], xv[j] * alpha);
  }
}

// ---------------- LayerNorm(a+b) : one wave per node ----------------
__global__ __launch_bounds__(128) void k_ln(const float* a, const float* b,
    const float* g, const float* bb, float* out) {
  int wave = threadIdx.x >> 5, lane = threadIdx.x & 31;
  int node = blockIdx.x * 4 + wave;
  if (node >= N_NODES) return;
  v4f va = *(const v4f*)(a + (size_t)node * DIM + lane * 4);
  v4f vb = *(const v4f*)(b + (size_t)node * DIM + lane * 4);
  v4f v = va + vb;
  float s = v[0] + v[1] + v[2] + v[3];
#pragma unroll
  for (int m = 16; m >= 1; m >>= 1) s += __shfl_xor(s, m, 32);
  float mu = s * (1.0f / DIM), vs = 0.f;
#pragma unroll
  for (int i = 0; i < 4; ++i) { float t = v[i] - mu; vs += t * t; }
#pragma unroll
  for (int m = 16; m >= 1; m >>= 1) vs += __shfl_xor(vs, m, 32);
  float rstd = rsqrtf(vs * (1.0f / DIM) + 1e-5f);
  v4f vg = *(const v4f*)(g + lane * 4);
  v4f vbb = *(const v4f*)(bb + lane * 4);
  v4f o;
#pragma unroll
  for (int i = 0; i < 4; ++i) o[i] = (v[i] - mu) * rstd * vg[i] + vbb[i];
  *(v4f*)(out + (size_t)node * DIM + lane * 4) = o;
}

// ---------------- MLP via WMMA ----------------
__global__ __launch_bounds__(128) void k_mlp1(const float* h,
    const unsigned short* W116, float* mid) {
  int wave = threadIdx.x >> 5, lane = threadIdx.x & 31;
  int ntile = blockIdx.x * 4 + wave;
  if (ntile * 16 >= N_NODES) return;
  int node = ntile * 16 + (lane & 15);
  int ksub = (lane >> 4) << 3;
  const float* hrow = h + (size_t)node * DIM;
  v16bf b[4];
#pragma unroll
  for (int kt = 0; kt < 4; ++kt) b[kt] = frag_from_f32(hrow, kt * 32 + ksub);
#pragma unroll
  for (int mt = 0; mt < 16; ++mt) {
    int mrow = mt * 16 + (lane & 15);
    v8f acc = {};
#pragma unroll
    for (int kt = 0; kt < 4; ++kt)
      acc = wmma_bf16(frag_from_u16(W116 + (size_t)mrow * DIM, kt * 32 + ksub), b[kt], acc);
    v8f ge;
#pragma unroll
    for (int i = 0; i < 8; ++i) {
      float xv = acc[i];
      ge[i] = 0.5f * xv * (1.0f + erff(xv * 0.70710678118f));  // exact gelu
    }
    store8(mid + (size_t)node * MID_DIM + mt * 16 + ksub, ge);
  }
}
__global__ __launch_bounds__(128) void k_mlp2(const float* mid,
    const unsigned short* W216, float* mlp) {
  int wave = threadIdx.x >> 5, lane = threadIdx.x & 31;
  int ntile = blockIdx.x * 4 + wave;
  if (ntile * 16 >= N_NODES) return;
  int node = ntile * 16 + (lane & 15);
  int ksub = (lane >> 4) << 3;
  const float* mrow_ = mid + (size_t)node * MID_DIM;
  v16bf b[8];
#pragma unroll
  for (int kt = 0; kt < 8; ++kt) b[kt] = frag_from_f32(mrow_, kt * 32 + ksub);
#pragma unroll
  for (int mt = 0; mt < 8; ++mt) {
    int mrow = mt * 16 + (lane & 15);
    v8f acc = {};
#pragma unroll
    for (int kt = 0; kt < 8; ++kt)
      acc = wmma_bf16(frag_from_u16(W216 + (size_t)mrow * MID_DIM, kt * 32 + ksub), b[kt], acc);
    store8(mlp + (size_t)node * DIM + mt * 16 + ksub, acc);
  }
}

// ---------------- launch ----------------
extern "C" void kernel_launch(void* const* d_in, const int* in_sizes, int n_in,
                              void* d_out, int out_size, void* d_ws, size_t ws_size,
                              hipStream_t stream) {
  const float* x     = (const float*)d_in[0];
  const int*   ei    = (const int*)d_in[1];
  const float* ea    = (const float*)d_in[2];
  const float* Wl    = (const float*)d_in[3];
  const float* bl    = (const float*)d_in[4];
  const float* Wr    = (const float*)d_in[5];
  const float* br    = (const float*)d_in[6];
  const float* We    = (const float*)d_in[7];
  const float* att   = (const float*)d_in[8];
  const float* gbias = (const float*)d_in[9];
  const float* ln1g  = (const float*)d_in[10];
  const float* ln1b  = (const float*)d_in[11];
  const float* ln2g  = (const float*)d_in[12];
  const float* ln2b  = (const float*)d_in[13];
  const float* W1    = (const float*)d_in[14];
  const float* W2    = (const float*)d_in[15];

  float* out       = (float*)d_out;                         // [N,128]
  int*   ei_out    = (int*)(out + (size_t)N_NODES * DIM);   // [2, EN]
  float* alpha_out = (float*)(ei_out + 2 * (size_t)EN);     // [EN, 4]

  char* ws = (char*)d_ws;
  float* deg   = (float*)(ws + OFF_DEG);
  float* loopb = (float*)(ws + OFF_LOOP);
  float* xl    = (float*)(ws + OFF_XL);
  float* xr    = (float*)(ws + OFF_XR);
  float* gat   = (float*)(ws + OFF_GAT);
  float* hbuf  = (float*)(ws + OFF_H);
  float* mid   = (float*)(ws + OFF_MID);
  float* mlp   = (float*)(ws + OFF_MLP);
  float* logits= (float*)(ws + OFF_LOG);
  float* amax  = (float*)(ws + OFF_AMAX);
  float* denom = (float*)(ws + OFF_DEN);
  unsigned short* Wl16 = (unsigned short*)(ws + OFF_WL16);
  unsigned short* Wr16 = (unsigned short*)(ws + OFF_WR16);
  unsigned short* We16 = (unsigned short*)(ws + OFF_WE16);
  unsigned short* W116 = (unsigned short*)(ws + OFF_W116);
  unsigned short* W216 = (unsigned short*)(ws + OFF_W216);

  auto cdiv = [](int a, int b) { return (a + b - 1) / b; };
  const float NEG_INF = -__builtin_inff();

  // init
  k_fill<<<cdiv(N_NODES, 256), 256, 0, stream>>>(deg, 0.f, N_NODES);
  k_fill<<<cdiv(N_NODES * EDIM, 256), 256, 0, stream>>>(loopb, 0.f, N_NODES * EDIM);
  k_fill<<<cdiv(N_NODES * HEADS, 256), 256, 0, stream>>>(denom, 0.f, N_NODES * HEADS);
  k_fill<<<cdiv(N_NODES * HEADS, 256), 256, 0, stream>>>(amax, NEG_INF, N_NODES * HEADS);
  k_init_gat<<<cdiv(N_NODES * DIM, 256), 256, 0, stream>>>(gat, gbias, N_NODES * DIM);

  // weights -> bf16
  k_cvt_bf16<<<cdiv(DIM * DIM, 256), 256, 0, stream>>>(Wl, Wl16, DIM * DIM);
  k_cvt_bf16<<<cdiv(DIM * DIM, 256), 256, 0, stream>>>(Wr, Wr16, DIM * DIM);
  k_cvt_bf16<<<cdiv(DIM * EDIM, 256), 256, 0, stream>>>(We, We16, DIM * EDIM);
  k_cvt_bf16<<<cdiv(MID_DIM * DIM, 256), 256, 0, stream>>>(W1, W116, MID_DIM * DIM);
  k_cvt_bf16<<<cdiv(DIM * MID_DIM, 256), 256, 0, stream>>>(W2, W216, DIM * MID_DIM);

  // edge index output (with self loops)
  k_ei<<<cdiv(EN, 256), 256, 0, stream>>>(ei, ei_out);

  // self-loop attrs
  k_loop_accum<<<cdiv(N_EDGES * 4, 256), 256, 0, stream>>>(ei, ea, deg, loopb);
  k_loop_div<<<cdiv(N_NODES * EDIM, 256), 256, 0, stream>>>(deg, loopb);

  // projections (WMMA)
  int ntiles = cdiv(N_NODES, 16);
  k_proj<<<cdiv(ntiles, 4), 128, 0, stream>>>(x, Wl16, bl, Wr16, br, xl, xr);

  // edge logits (WMMA + LDS) and segment max
  int etiles = cdiv(EN, 16);
  k_edge<<<cdiv(etiles, 4), 128, 0, stream>>>(ei, ea, loopb, We16, xl, xr, att,
                                              logits, amax);

  // softmax + aggregation
  k_softmax<<<cdiv(EN * HEADS, 256), 256, 0, stream>>>(ei, logits, amax, denom, alpha_out);
  k_agg<<<cdiv(EN * HEADS, 256), 256, 0, stream>>>(ei, denom, xl, alpha_out, gat);

  // LN1, MLP (WMMA), LN2
  k_ln<<<cdiv(N_NODES, 4), 128, 0, stream>>>(x, gat, ln1g, ln1b, hbuf);
  k_mlp1<<<cdiv(ntiles, 4), 128, 0, stream>>>(hbuf, W116, mid);
  k_mlp2<<<cdiv(ntiles, 4), 128, 0, stream>>>(mid, W216, mlp);
  k_ln<<<cdiv(N_NODES, 4), 128, 0, stream>>>(hbuf, mlp, ln2g, ln2b, out);
}
